// AdaptiveVoxelizationGPU_83537113907662
// MI455X (gfx1250) — matrix operand order repaired and checked by reading, not compiled, not run on hardware
//
#include <hip/hip_runtime.h>
#include <cstdint>

#define BLK 256
#define PPT 16
#define SCAN_PTS (BLK * PPT)   // 4096 points per scan block
#define INS_ITERS 8            // 2048 points per insert block
#define MAXP 35
#define SENT 0x7fffffff

struct VoxParam {
  float ox, oy, oz;
  float vx, vy, vz;
  int nx, ny, nz;
};

typedef __attribute__((ext_vector_type(16))) _Float16 v16h;
typedef __attribute__((ext_vector_type(8)))  float    v8f;

// ---------------- CDNA5 async global->LDS copy (ASYNCcnt path) ----------------
__device__ __forceinline__ void async_load_b128(unsigned lds_off, const void* gptr) {
  asm volatile("global_load_async_to_lds_b128 %0, %1, off"
               :: "v"(lds_off), "v"(gptr)
               : "memory");
}
__device__ __forceinline__ void wait_async_le1() {
  asm volatile("s_wait_asynccnt 0x1" ::: "memory");
}
__device__ __forceinline__ void wait_async_le0() {
  asm volatile("s_wait_asynccnt 0x0" ::: "memory");
}

// ---------------- voxel coordinate helpers ----------------
__device__ __forceinline__ int voxel_of(float4 p, const VoxParam P) {
  float fx = floorf((p.x - P.ox) / P.vx);
  float fy = floorf((p.y - P.oy) / P.vy);
  float fz = floorf((p.z - P.oz) / P.vz);
  int cx = (int)fx, cy = (int)fy, cz = (int)fz;
  if (cx < 0 || cx >= P.nx || cy < 0 || cy >= P.ny || cz < 0 || cz >= P.nz) return -1;
  return (cz * P.ny + cy) * P.nx + cx;
}

// ---------------- WMMA wave32 inclusive scan ----------------
// D[m][n] = sum_k A[m][k]*B[k][n].  A0[m][k]=[k<=2m], B0[k][n]=x_k  -> P(2m)
// A1[m][k]=[k==2m+1], B1[k][n]=x_k*[n==1]                          -> +x_{2m+1} at n==1
// lane l reads D[l>>1][l&1] = inclusive prefix P(l).  Exact: x_k<=16 (f16 int),
// accumulation in f32.
__device__ __forceinline__ int wave_incl_scan_wmma(int sval) {
  const int lane = threadIdx.x & 31;
  const int m = lane & 15;
  const bool hi = lane >= 16;
  float vf = (float)sval;
  v16h a0, a1, b0, b1;
#pragma unroll
  for (int h = 0; h < 16; ++h) {
    // A-matrix 16x32 f16 layout: lane<16: K = h (h<8) else h+8 ; lane>=16: K = h+8 (h<8) else h+16
    int kA = hi ? ((h < 8) ? (8 + h) : (16 + h)) : ((h < 8) ? h : (8 + h));
    // B-matrix 32x16 f16 layout: same K as A for element h, column N = lane&15
    int kB = kA;
    a0[h] = (kA <= 2 * m)     ? (_Float16)1.0f : (_Float16)0.0f;
    a1[h] = (kA == 2 * m + 1) ? (_Float16)1.0f : (_Float16)0.0f;
    float xv = __shfl(vf, kB, 32);
    b0[h] = (_Float16)xv;
    b1[h] = (m == 1) ? (_Float16)xv : (_Float16)0.0f;
  }
  v8f c = {0.f, 0.f, 0.f, 0.f, 0.f, 0.f, 0.f, 0.f};
  c = __builtin_amdgcn_wmma_f32_16x16x32_f16(false, a0, false, b0, (short)0, c, false, false);
  c = __builtin_amdgcn_wmma_f32_16x16x32_f16(false, a1, false, b1, (short)0, c, false, false);
  // element (M=l>>1, N=l&1) lives at lane (N + 16*[M>=8]), VGPR M&7
  int src = (lane & 1) + (((lane >> 1) >= 8) ? 16 : 0);
  int rsel = (lane >> 1) & 7;
  float pfx = 0.f;
#pragma unroll
  for (int r = 0; r < 8; ++r) {
    float t = __shfl(c[r], src, 32);
    if (rsel == r) pfx = t;
  }
  return (int)(pfx + 0.5f);
}

// ---------------- kernels ----------------
__global__ void fill_i32(int* __restrict__ p, long n, int v) {
  long i = (long)blockIdx.x * blockDim.x + threadIdx.x;
  long stride = (long)gridDim.x * blockDim.x;
  for (; i < n; i += stride) p[i] = v;
}

// Per-point: compute voxel id, store lin[i], insert point index into the
// per-voxel 35-slot min-cascade (keeps the 35 smallest indices, sorted).
__global__ __launch_bounds__(BLK)
void insert_kernel(const float4* __restrict__ pts, int N,
                   int* __restrict__ slotIdx, int* __restrict__ lin, VoxParam P) {
  __shared__ float4 buf[2][BLK];
  const int tid = threadIdx.x;
  const int base = blockIdx.x * (BLK * INS_ITERS);
  {
    int i0 = base + tid; if (i0 > N - 1) i0 = N - 1;
    async_load_b128((unsigned)(size_t)&buf[0][tid], pts + i0);
  }
  for (int it = 0; it < INS_ITERS; ++it) {
    if (it + 1 < INS_ITERS) {
      int i1 = base + (it + 1) * BLK + tid; if (i1 > N - 1) i1 = N - 1;
      async_load_b128((unsigned)(size_t)&buf[(it + 1) & 1][tid], pts + i1);
      wait_async_le1();
    } else {
      wait_async_le0();
    }
    const int i = base + it * BLK + tid;
    float4 p = buf[it & 1][tid];
    int g = -1;
    if (i < N) {
      g = voxel_of(p, P);
      lin[i] = g;
    }
    if (g >= 0) {
      int* row = slotIdx + (size_t)g * MAXP;
      int key = i;
#pragma unroll 1
      for (int s = 0; s < MAXP; ++s) {
        int v = row[s];                 // monotonic-decreasing: stale read is safe to skip on
        if (v < key) continue;
        int old = atomicMin(&row[s], key);
        if (old > key) {                // inserted key; carry displaced value forward
          if (old == SENT) break;       // consumed sentinel: done
          key = old;
        }
        // old < key: slot kept smaller value, retry same key at next slot
      }
    }
  }
}

// Block partial counts of "new voxel" flags (point order preserved by partition).
__global__ __launch_bounds__(BLK)
void scan1_kernel(const int* __restrict__ lin, const int* __restrict__ slotIdx,
                  int N, int* __restrict__ blockSums) {
  __shared__ int red[BLK];
  const int t = threadIdx.x;
  const int base = blockIdx.x * SCAN_PTS;
  int s = 0;
#pragma unroll
  for (int j = 0; j < PPT; ++j) {
    int i = base + t * PPT + j;
    if (i < N) {
      int g = lin[i];
      if (g >= 0 && slotIdx[(size_t)g * MAXP] == i) ++s;
    }
  }
  red[t] = s;
  __syncthreads();
  for (int off = BLK / 2; off > 0; off >>= 1) {
    if (t < off) red[t] += red[t + off];
    __syncthreads();
  }
  if (t == 0) blockSums[blockIdx.x] = red[0];
}

// Single-block exclusive scan of block partials; emits total and k output.
__global__ __launch_bounds__(1024)
void scan2_kernel(int* __restrict__ bs, int nb, int* __restrict__ total,
                  float* __restrict__ kOut, int maxv) {
  __shared__ int sh[1024];
  __shared__ int carry;
  const int t = threadIdx.x;
  if (t == 0) carry = 0;
  __syncthreads();
  for (int b = 0; b < nb; b += 1024) {
    int idx = b + t;
    int v = (idx < nb) ? bs[idx] : 0;
    sh[t] = v;
    __syncthreads();
    for (int off = 1; off < 1024; off <<= 1) {
      int add = (t >= off) ? sh[t - off] : 0;
      __syncthreads();
      sh[t] += add;
      __syncthreads();
    }
    int incl = sh[t] + carry;
    if (idx < nb) bs[idx] = incl - v;   // exclusive
    __syncthreads();
    if (t == 1023) carry = incl;
    __syncthreads();
  }
  if (t == 0) {
    int tt = carry;
    *total = tt;
    *kOut = (float)((tt < maxv) ? tt : maxv);
  }
}

// Assign slots (first-occurrence order), record slot->voxel map and coords (z,y,x).
__global__ __launch_bounds__(BLK)
void scan3_kernel(const int* __restrict__ lin, const int* __restrict__ slotIdx,
                  const float4* __restrict__ pts, int N,
                  const int* __restrict__ blockOffs, int* __restrict__ slot2g,
                  float* __restrict__ coors, int maxv, VoxParam P) {
  __shared__ int wtot[BLK / 32];
  const int t = threadIdx.x, wid = t >> 5, lane = t & 31;
  const int base = blockIdx.x * SCAN_PTS;
  unsigned mask = 0;
#pragma unroll
  for (int j = 0; j < PPT; ++j) {
    int i = base + t * PPT + j;
    int f = 0;
    if (i < N) {
      int g = lin[i];
      f = (g >= 0 && slotIdx[(size_t)g * MAXP] == i) ? 1 : 0;
    }
    mask |= (unsigned)f << j;
  }
  int s = __popc(mask);
  int incl = wave_incl_scan_wmma(s);        // v_wmma path
  if (lane == 31) wtot[wid] = incl;
  __syncthreads();
  int woff = 0;
  for (int w = 0; w < wid; ++w) woff += wtot[w];
  int run = blockOffs[blockIdx.x] + woff + (incl - s);
#pragma unroll
  for (int j = 0; j < PPT; ++j) {
    if ((mask >> j) & 1u) {
      if (run < maxv) {
        int i = base + t * PPT + j;
        int g = lin[i];
        slot2g[run] = g;
        float4 p = pts[i];
        int cx = (int)floorf((p.x - P.ox) / P.vx);
        int cy = (int)floorf((p.y - P.oy) / P.vy);
        int cz = (int)floorf((p.z - P.oz) / P.vz);
        coors[(size_t)run * 3 + 0] = (float)cz;
        coors[(size_t)run * 3 + 1] = (float)cy;
        coors[(size_t)run * 3 + 2] = (float)cx;
      }
      ++run;
    }
  }
}

// Gather kept voxel rows; zero-fill everything else (d_out is poisoned).
__global__ __launch_bounds__(BLK)
void gather_kernel(const float4* __restrict__ pts, const int* __restrict__ slotIdx,
                   const int* __restrict__ slot2g, const int* __restrict__ totalPtr,
                   int maxv, int N, float* __restrict__ voxOut,
                   float* __restrict__ coorsOut, float* __restrict__ numOut) {
  int v = blockIdx.x * blockDim.x + threadIdx.x;
  if (v >= maxv) return;
  int K = *totalPtr;
  if (K > maxv) K = maxv;
  float* row = voxOut + (size_t)v * MAXP * 4;
  if (v < K) {
    const int* si = slotIdx + (size_t)slot2g[v] * MAXP;
    int cnt = 0;
#pragma unroll 1
    for (int s = 0; s < MAXP; ++s) {
      int pi = si[s];
      if (pi != SENT) {
        float4 p = pts[pi];
        row[s * 4 + 0] = p.x; row[s * 4 + 1] = p.y;
        row[s * 4 + 2] = p.z; row[s * 4 + 3] = p.w;
        ++cnt;
      } else {
        row[s * 4 + 0] = 0.f; row[s * 4 + 1] = 0.f;
        row[s * 4 + 2] = 0.f; row[s * 4 + 3] = 0.f;
      }
    }
    numOut[v] = (float)cnt;
  } else {
#pragma unroll 1
    for (int s = 0; s < MAXP * 4; ++s) row[s] = 0.f;
    numOut[v] = 0.f;
    coorsOut[(size_t)v * 3 + 0] = 0.f;
    coorsOut[(size_t)v * 3 + 1] = 0.f;
    coorsOut[(size_t)v * 3 + 2] = 0.f;
  }
}

// ---------------- host side ----------------
static inline size_t align256(size_t x) { return (x + 255) & ~(size_t)255; }

static void run_one(const float4* pts, int N, VoxParam P, int G, int maxv,
                    float* voxOut, float* coorsOut, float* numOut, float* kOut,
                    uint8_t* ws, size_t& cur, hipStream_t stream) {
  int* slotIdx = (int*)(ws + cur); cur = align256(cur + (size_t)G * MAXP * 4);
  int* lin     = (int*)(ws + cur); cur = align256(cur + (size_t)N * 4);
  const int nblk = (N + SCAN_PTS - 1) / SCAN_PTS;
  int* bsums   = (int*)(ws + cur); cur = align256(cur + (size_t)nblk * 4);
  int* slot2g  = (int*)(ws + cur); cur = align256(cur + (size_t)maxv * 4);
  int* total   = (int*)(ws + cur); cur = align256(cur + 256);

  fill_i32<<<1024, BLK, 0, stream>>>(slotIdx, (long)G * MAXP, SENT);
  const int insBlocks = (N + BLK * INS_ITERS - 1) / (BLK * INS_ITERS);
  insert_kernel<<<insBlocks, BLK, 0, stream>>>(pts, N, slotIdx, lin, P);
  scan1_kernel<<<nblk, BLK, 0, stream>>>(lin, slotIdx, N, bsums);
  scan2_kernel<<<1, 1024, 0, stream>>>(bsums, nblk, total, kOut, maxv);
  scan3_kernel<<<nblk, BLK, 0, stream>>>(lin, slotIdx, pts, N, bsums, slot2g,
                                         coorsOut, maxv, P);
  gather_kernel<<<(maxv + BLK - 1) / BLK, BLK, 0, stream>>>(
      pts, slotIdx, slot2g, total, maxv, N, voxOut, coorsOut, numOut);
}

extern "C" void kernel_launch(void* const* d_in, const int* in_sizes, int n_in,
                              void* d_out, int out_size, void* d_ws, size_t ws_size,
                              hipStream_t stream) {
  const float4* pts = (const float4*)d_in[0];
  const int N = in_sizes[0] / 4;
  float* out = (float*)d_out;

  const int MAXVD = 12000, MAXVS = 6000;
  size_t o = 0;
  float* vd = out + o; o += (size_t)MAXVD * MAXP * 4;
  float* cd = out + o; o += (size_t)MAXVD * 3;
  float* nd = out + o; o += (size_t)MAXVD;
  float* kd = out + o; o += 1;
  float* vs = out + o; o += (size_t)MAXVS * MAXP * 4;
  float* cs = out + o; o += (size_t)MAXVS * 3;
  float* ns = out + o; o += (size_t)MAXVS;
  float* ks = out + o; o += 1;

  VoxParam Pd{0.f, -39.68f, -3.f, 0.16f, 0.16f, 4.f, 432, 496, 1};
  VoxParam Ps{0.f, -39.68f, -3.f, 0.32f, 0.32f, 4.f, 216, 248, 1};

  size_t cur = 0;
  run_one(pts, N, Pd, 432 * 496 * 1, MAXVD, vd, cd, nd, kd,
          (uint8_t*)d_ws, cur, stream);
  run_one(pts, N, Ps, 216 * 248 * 1, MAXVS, vs, cs, ns, ks,
          (uint8_t*)d_ws, cur, stream);
}
